// FeaturePropagation_23115513987445
// MI455X (gfx1250) — matrix operand order repaired and checked by reading, not compiled, not run on hardware
//
#include <hip/hip_runtime.h>
#include <hip/hip_bf16.h>

// ---------------------------------------------------------------- constants
#define NBATCH     4
#define N1_PER     16384
#define N2_PER     2048
#define N1_TOTAL   (NBATCH * N1_PER)   // 65536
#define CFINE      64
#define CCOARSE    128
#define C0         192                 // CFINE + CCOARSE
#define C1         256
#define C2         128
#define EPS_ID     1e-8f
#define LN_EPS     1e-5f

typedef __attribute__((ext_vector_type(16))) __bf16 v16bf;
typedef __attribute__((ext_vector_type(8)))  __bf16 v8bf;
typedef __attribute__((ext_vector_type(8)))  float  v8f;

__device__ __forceinline__ unsigned int pack_bf16(float a, float b) {
    unsigned short lo = __builtin_bit_cast(unsigned short, (__bf16)a);
    unsigned short hi = __builtin_bit_cast(unsigned short, (__bf16)b);
    return (unsigned int)lo | ((unsigned int)hi << 16);
}

// -------------------------------------------------- WMMA fragment loaders
// A (16-bit 16x32, MxK): lane L -> row M = L%16.
//   lanes 0-15 : elems 0..7 = K k0+0..7,  elems 8..15 = K k0+16..23
//   lanes 16-31: elems 0..7 = K k0+8..15, elems 8..15 = K k0+24..31
__device__ __forceinline__ v16bf load_a_frag(const __bf16* aT, int lane, int k0, int lda) {
    int m  = lane & 15;
    int kb = k0 + ((lane & 16) ? 8 : 0);
    const __bf16* p = aT + m * lda + kb;
    v8bf lo = *reinterpret_cast<const v8bf*>(p);
    v8bf hi = *reinterpret_cast<const v8bf*>(p + 16);
    return __builtin_shufflevector(lo, hi, 0,1,2,3,4,5,6,7,8,9,10,11,12,13,14,15);
}

// B (16-bit 32x16, KxN) from W^T stored [N][K] row-major: lane L -> col N = n0 + L%16.
//   lanes 0-15: elems 0..7 = K k0+0..7, elems 8..15 = K k0+16..23; lanes 16-31: K k0+8..15, k0+24..31.
__device__ __forceinline__ v16bf load_b_frag(const __bf16* wT, int lane, int n0, int k0, int K) {
    int n  = n0 + (lane & 15);
    int kb = k0 + ((lane & 16) ? 8 : 0);
    const __bf16* p = wT + (size_t)n * K + kb;
    v8bf lo = *reinterpret_cast<const v8bf*>(p);
    v8bf hi = *reinterpret_cast<const v8bf*>(p + 16);
    return __builtin_shufflevector(lo, hi, 0,1,2,3,4,5,6,7,8,9,10,11,12,13,14,15);
}

// ------------------------------------------------ weight convert+transpose
__global__ void fp_wcvt_kernel(const float* __restrict__ W1, const float* __restrict__ W2,
                               __bf16* __restrict__ W1T, __bf16* __restrict__ W2T) {
    int tid = blockIdx.x * blockDim.x + threadIdx.x;
    if (tid < C0 * C1) {                       // W1 [C0,C1] -> W1T [C1][C0]
        int k = tid / C1, n = tid % C1;
        W1T[n * C0 + k] = (__bf16)W1[tid];
    } else {
        int t = tid - C0 * C1;
        if (t < C1 * C2) {                     // W2 [C1,C2] -> W2T [C2][C1]
            int k = t / C2, n = t % C2;
            W2T[n * C1 + k] = (__bf16)W2[t];
        }
    }
}

// ------------------------------------- 3-NN inverse-distance interpolation
// Writes concatenated activation row [feature1(64) | interp(128)] as packed bf16.
__global__ __launch_bounds__(256)
void fp_interp_kernel(const float* __restrict__ xyz1, const float* __restrict__ xyz2,
                      const float* __restrict__ f1,   const float* __restrict__ f2,
                      __bf16* __restrict__ xcat) {
    __shared__ float sx[N2_PER], sy[N2_PER], sz[N2_PER];   // 24 KB: whole coarse cloud of this batch
    const int blocksPerBatch = N1_PER / 256;
    const int b = blockIdx.x / blocksPerBatch;
    const int q = b * N1_PER + (blockIdx.x % blocksPerBatch) * 256 + threadIdx.x;

    const float* src = xyz2 + (size_t)b * N2_PER * 3;
    for (int i = threadIdx.x; i < N2_PER; i += 256) {
        sx[i] = src[3 * i + 0];
        sy[i] = src[3 * i + 1];
        sz[i] = src[3 * i + 2];
    }
    __syncthreads();

    const float qx = xyz1[3 * q + 0], qy = xyz1[3 * q + 1], qz = xyz1[3 * q + 2];
    float d0 = 3.4e38f, d1 = 3.4e38f, d2 = 3.4e38f;
    int   i0 = 0, i1 = 0, i2 = 0;
    // Branchless top-3 insertion (v_cmp + v_cndmask); unroll x4 so the LDS
    // reads of consecutive j merge into ds_load_b128.
    #pragma unroll 4
    for (int j = 0; j < N2_PER; ++j) {
        float dx = sx[j] - qx, dy = sy[j] - qy, dz = sz[j] - qz;
        float d  = dx * dx + dy * dy + dz * dz;
        bool c0 = d < d0, c1 = d < d1, c2 = d < d2;
        d2 = c1 ? d1 : (c2 ? d : d2);  i2 = c1 ? i1 : (c2 ? j : i2);
        d1 = c0 ? d0 : (c1 ? d : d1);  i1 = c0 ? i0 : (c1 ? j : i1);
        d0 = c0 ? d  : d0;             i0 = c0 ? j  : i0;
    }
    float r0 = 1.0f / (sqrtf(fmaxf(d0, 0.0f)) + EPS_ID);
    float r1 = 1.0f / (sqrtf(fmaxf(d1, 0.0f)) + EPS_ID);
    float r2 = 1.0f / (sqrtf(fmaxf(d2, 0.0f)) + EPS_ID);
    float rs = 1.0f / (r0 + r1 + r2);
    float w0 = r0 * rs, w1 = r1 * rs, w2 = r2 * rs;

    // Packed dword stores: row base q*C0*2 bytes is 384-aligned.
    unsigned int* outw = reinterpret_cast<unsigned int*>(xcat + (size_t)q * C0);
    const float4* f1v = reinterpret_cast<const float4*>(f1 + (size_t)q * CFINE);
    #pragma unroll
    for (int c = 0; c < CFINE / 4; ++c) {
        float4 v = f1v[c];
        outw[2 * c + 0] = pack_bf16(v.x, v.y);
        outw[2 * c + 1] = pack_bf16(v.z, v.w);
    }

    const float* fb = f2 + (size_t)b * N2_PER * CCOARSE;
    const float4* p0 = reinterpret_cast<const float4*>(fb + (size_t)i0 * CCOARSE);
    const float4* p1 = reinterpret_cast<const float4*>(fb + (size_t)i1 * CCOARSE);
    const float4* p2 = reinterpret_cast<const float4*>(fb + (size_t)i2 * CCOARSE);
    #pragma unroll 4
    for (int c = 0; c < CCOARSE / 4; ++c) {
        float4 a = p0[c], bb = p1[c], d = p2[c];
        float x = w0 * a.x + w1 * bb.x + w2 * d.x;
        float y = w0 * a.y + w1 * bb.y + w2 * d.y;
        float z = w0 * a.z + w1 * bb.z + w2 * d.z;
        float w = w0 * a.w + w1 * bb.w + w2 * d.w;
        outw[CFINE / 2 + 2 * c + 0] = pack_bf16(x, y);
        outw[CFINE / 2 + 2 * c + 1] = pack_bf16(z, w);
    }
}

// ----------------------------- fused WMMA GEMM + bias + LayerNorm + ReLU
// One block (8 waves) owns 16 rows x all NOUT columns; each wave does TPW
// 16x16 tiles (8*TPW*16 == NOUT). LayerNorm reduces within the block via LDS.
template <int K, int NOUT, int TPW>
__global__ __launch_bounds__(256)
void fp_gemm_ln_relu_kernel(const __bf16* __restrict__ A, const __bf16* __restrict__ WT,
                            const float* __restrict__ bias, const float* __restrict__ gamma,
                            const float* __restrict__ beta,
                            __bf16* __restrict__ outB, float* __restrict__ outF) {
    __shared__ __align__(16) __bf16 aT[16 * K];
    __shared__ float hOut[16][NOUT];
    __shared__ float rsum[16][16];
    __shared__ float rsum2[16][16];
    __shared__ float rowMu[16];
    __shared__ float rowRs[16];

    const int tid  = threadIdx.x;
    const int row0 = blockIdx.x * 16;

    // A tile: 16 consecutive rows of [*, K] -> one contiguous chunk of 16*K bf16
    {
        const unsigned int* gA = reinterpret_cast<const unsigned int*>(A + (size_t)row0 * K);
        unsigned int* sA = reinterpret_cast<unsigned int*>(aT);
        #pragma unroll
        for (int i = tid; i < 16 * K / 2; i += 256) sA[i] = gA[i];
    }
    __syncthreads();

    const int wave = tid >> 5;
    const int lane = tid & 31;

    v8f acc[TPW];
    #pragma unroll
    for (int t = 0; t < TPW; ++t)
        #pragma unroll
        for (int v = 0; v < 8; ++v) acc[t][v] = 0.0f;

    #pragma unroll
    for (int k0 = 0; k0 < K; k0 += 32) {
        v16bf af = load_a_frag(aT, lane, k0, K);
        #pragma unroll
        for (int t = 0; t < TPW; ++t) {
            int n0 = (wave * TPW + t) * 16;
            v16bf bf = load_b_frag(WT, lane, n0, k0, K);
            acc[t] = __builtin_amdgcn_wmma_f32_16x16x32_bf16(
                false, af, false, bf, (short)0, acc[t], false, false);
        }
    }

    // D layout: elem v -> row = v + 8*(lane>=16), col = n0 + lane%16
    const int rowAdd = (lane & 16) ? 8 : 0;
    const int colL   = lane & 15;
    #pragma unroll
    for (int t = 0; t < TPW; ++t) {
        int col = (wave * TPW + t) * 16 + colL;
        float bv = bias[col];
        #pragma unroll
        for (int v = 0; v < 8; ++v) hOut[v + rowAdd][col] = acc[t][v] + bv;
    }
    __syncthreads();

    // LayerNorm reduction: 16 threads per row
    {
        const int CPS = NOUT / 16;
        int r = tid >> 4, seg = tid & 15;
        float s = 0.0f, s2 = 0.0f;
        #pragma unroll
        for (int i = 0; i < CPS; ++i) {
            float v = hOut[r][seg * CPS + i];
            s += v; s2 += v * v;
        }
        rsum[r][seg] = s; rsum2[r][seg] = s2;
    }
    __syncthreads();
    if (tid < 16) {
        float s = 0.0f, s2 = 0.0f;
        #pragma unroll
        for (int i = 0; i < 16; ++i) { s += rsum[tid][i]; s2 += rsum2[tid][i]; }
        float mu  = s / (float)NOUT;
        float var = s2 / (float)NOUT - mu * mu;
        rowMu[tid] = mu;
        rowRs[tid] = rsqrtf(var + LN_EPS);
    }
    __syncthreads();

    // scale/shift + ReLU + store
    #pragma unroll
    for (int idx = tid; idx < 16 * NOUT; idx += 256) {
        int r = idx / NOUT, c = idx % NOUT;
        float v = (hOut[r][c] - rowMu[r]) * rowRs[r] * gamma[c] + beta[c];
        v = fmaxf(v, 0.0f);
        size_t o = (size_t)(row0 + r) * NOUT + c;
        if (outF) outF[o] = v;
        else      outB[o] = (__bf16)v;
    }
}

// ------------------------------------------------------------------ launch
extern "C" void kernel_launch(void* const* d_in, const int* in_sizes, int n_in,
                              void* d_out, int out_size, void* d_ws, size_t ws_size,
                              hipStream_t stream) {
    (void)in_sizes; (void)n_in; (void)out_size; (void)ws_size;
    const float* xyz1 = (const float*)d_in[0];
    const float* xyz2 = (const float*)d_in[1];
    const float* f1   = (const float*)d_in[2];
    const float* f2   = (const float*)d_in[3];
    // d_in[4], d_in[5] = offset1/offset2 (equal-sized batches; compile-time constants used)
    const float* W1   = (const float*)d_in[6];
    const float* b1   = (const float*)d_in[7];
    const float* g1   = (const float*)d_in[8];
    const float* be1  = (const float*)d_in[9];
    const float* W2   = (const float*)d_in[10];
    const float* b2   = (const float*)d_in[11];
    const float* g2   = (const float*)d_in[12];
    const float* be2  = (const float*)d_in[13];

    char* ws = (char*)d_ws;
    const size_t szXcat = (size_t)N1_TOTAL * C0 * sizeof(__bf16);   // 25,165,824
    const size_t szH1   = (size_t)N1_TOTAL * C1 * sizeof(__bf16);   // 33,554,432
    const size_t szW1T  = (size_t)C1 * C0 * sizeof(__bf16);         //     98,304
    __bf16* xcat = (__bf16*)(ws);
    __bf16* h1   = (__bf16*)(ws + szXcat);
    __bf16* W1T  = (__bf16*)(ws + szXcat + szH1);
    __bf16* W2T  = (__bf16*)(ws + szXcat + szH1 + szW1T);
    float*  out  = (float*)d_out;

    fp_wcvt_kernel<<<(C0 * C1 + C1 * C2 + 255) / 256, 256, 0, stream>>>(W1, W2, W1T, W2T);
    fp_interp_kernel<<<N1_TOTAL / 256, 256, 0, stream>>>(xyz1, xyz2, f1, f2, xcat);
    fp_gemm_ln_relu_kernel<C0, C1, 2><<<N1_TOTAL / 16, 256, 0, stream>>>(
        xcat, W1T, b1, g1, be1, h1, nullptr);
    fp_gemm_ln_relu_kernel<C1, C2, 1><<<N1_TOTAL / 16, 256, 0, stream>>>(
        h1, W2T, b2, g2, be2, nullptr, out);
}
